// Rope_MHA_84482006713329
// MI455X (gfx1250) — matrix operand order repaired and checked by reading, not compiled, not run on hardware
//
#include <hip/hip_runtime.h>
#include <hip/hip_bf16.h>

typedef __attribute__((ext_vector_type(16))) _Float16 v16h;
typedef __attribute__((ext_vector_type(8)))  _Float16 v8h;
typedef __attribute__((ext_vector_type(4)))  _Float16 v4h;
typedef __attribute__((ext_vector_type(8)))  float    v8f;

__device__ __forceinline__ v16h pack16(v8h lo, v8h hi) {
  union { v16h v; v8h h[2]; } u;
  u.h[0] = lo; u.h[1] = hi;
  return u.v;
}

__device__ __forceinline__ v8f wmma_f16(v16h a, v16h b, v8f c) {
  // D = A(16x32 f16) * B(32x16 f16) + C(16x16 f32)
  return __builtin_amdgcn_wmma_f32_16x16x32_f16(false, a, false, b, (short)0, c,
                                                false, false);
}

// CDNA5 async global->LDS copy, 16B per lane (tracked by ASYNCcnt).
// Generic LDS pointer's low 32 bits are the LDS byte offset (ISA 10.2 aperture).
__device__ __forceinline__ void async_copy16(const _Float16* lds_dst,
                                             const _Float16* gsrc) {
  unsigned int ldsa = (unsigned int)(unsigned long long)lds_dst;
  unsigned long long ga = (unsigned long long)gsrc;
  asm volatile("global_load_async_to_lds_b128 %0, %1, off"
               :: "v"(ldsa), "v"(ga) : "memory");
}
__device__ __forceinline__ void async_wait0() {
  asm volatile("s_wait_asynccnt 0x0" ::: "memory");
}

// ---------------------------------------------------------------- f32 -> f16
__global__ __launch_bounds__(256) void k_cvt(const float* __restrict__ in,
                                             _Float16* __restrict__ out, int n) {
  int i = (blockIdx.x * 256 + threadIdx.x) * 4;
  if (i + 3 < n) {
    float4 f = *(const float4*)(in + i);
    v4h h;
    h[0] = (_Float16)f.x; h[1] = (_Float16)f.y;
    h[2] = (_Float16)f.z; h[3] = (_Float16)f.w;
    *(v4h*)(out + i) = h;
  }
}

// ------------------------------------------------- C[M,N] = A[M,K] * B[N,K]^T
// A, B: f16 row-major. C: f32. N, K compile-time for immediate-offset stores.
#define GBM 128
#define GBN 64
#define GBK 32
#define GLDA 40
#define GLDB 40

template <int N, int K>
__global__ __launch_bounds__(256) void k_gemm(const _Float16* __restrict__ A,
                                              const _Float16* __restrict__ B,
                                              float* __restrict__ C) {
  __shared__ __align__(16) _Float16 As[GBM * GLDA];
  __shared__ __align__(16) _Float16 Bs[GBN * GLDB];

  const int tid  = threadIdx.x;
  const int lane = tid & 31;
  const int wave = tid >> 5;
  const int wm   = wave >> 1;   // 0..3 (M strip of 32)
  const int wn   = wave & 1;    // 0..1 (N strip of 32)
  const int l15  = lane & 15;
  const int lh   = lane >> 4;
  const int bm   = blockIdx.x * GBM;
  const int bn   = blockIdx.y * GBN;

  v8f acc[2][2] = {};

  // A tile: 128 rows x 32 halves (64B/row) -> 2 x 16B chunks per thread
  const int arow = tid >> 1;
  const int acol = (tid & 1) * 16;
  // B tile: 64 rows x 32 halves (64B/row) -> 1 x 16B chunk per thread
  const int brow = tid >> 2;
  const int bcol = (tid & 3) * 8;

  for (int k0 = 0; k0 < K; k0 += GBK) {
    // ---- async global -> LDS (no VGPR round-trip)
    const _Float16* ga = A + (size_t)(bm + arow) * K + k0 + acol;
    async_copy16(&As[arow * GLDA + acol],     ga);
    async_copy16(&As[arow * GLDA + acol + 8], ga + 8);
    async_copy16(&Bs[brow * GLDB + bcol],
                 B + (size_t)(bn + brow) * K + k0 + bcol);
    async_wait0();
    __syncthreads();

    // ---- fragments (ISA VGPR layouts)
    v16h af[2], bf[2];
#pragma unroll
    for (int i = 0; i < 2; ++i) {
      // A 16x32: lane m = l&15; lo: k = 8*lh + e; hi: k = 16 + 8*lh + e
      const _Float16* p = &As[(wm * 32 + i * 16 + l15) * GLDA + lh * 8];
      af[i] = pack16(*(const v8h*)p, *(const v8h*)(p + 16));
    }
#pragma unroll
    for (int j = 0; j < 2; ++j) {
      // B 32x16: lane n = l&15; lo: k = 8*lh + e; hi: k = 16 + 8*lh + e
      const _Float16* p = &Bs[(wn * 32 + j * 16 + l15) * GLDB + lh * 8];
      bf[j] = pack16(*(const v8h*)p, *(const v8h*)(p + 16));
    }

#pragma unroll
    for (int i = 0; i < 2; ++i)
#pragma unroll
      for (int j = 0; j < 2; ++j)
        acc[i][j] = wmma_f16(af[i], bf[j], acc[i][j]);

    __syncthreads();
  }

  // ---- epilogue: C layout lane: n = l&15, m = v + 8*(l>>4)
  float* crow = C + (size_t)(bm + wm * 32 + lh * 8) * N + bn + wn * 32 + l15;
#pragma unroll
  for (int i = 0; i < 2; ++i)
#pragma unroll
    for (int j = 0; j < 2; ++j)
#pragma unroll
      for (int v = 0; v < 8; ++v)
        crow[(size_t)(i * 16 + v) * N + j * 16] = acc[i][j][v];
}

// -------------------------------------------- RoPE + split QKV + emit k,v f32
__global__ __launch_bounds__(256) void k_rope(const float* __restrict__ QKV,
                                              _Float16* __restrict__ qh,
                                              _Float16* __restrict__ kh,
                                              _Float16* __restrict__ vh,
                                              float* __restrict__ kout,
                                              float* __restrict__ vout,
                                              int S, int H) {
  const int lane = threadIdx.x & 31;
  const int r = blockIdx.x * 8 + (threadIdx.x >> 5);  // row in [0, B*H*S)
  const int s   = r % S;
  const int bhh = r / S;
  const int h   = bhh % H;
  const int b   = bhh / H;
  const float* src = QKV + (size_t)(b * S + s) * 3072;
  const int d = lane;  // this lane handles d and d+32 (rotate-half partner)

  float q1 = src[h * 64 + d],        q2 = src[h * 64 + d + 32];
  float k1 = src[1024 + h * 64 + d], k2 = src[1024 + h * 64 + d + 32];
  float v1 = src[2048 + h * 64 + d], v2 = src[2048 + h * 64 + d + 32];

  // freqs[j] = theta^{-j/32};  ln(10000)/32 = 0.28782313662...
  float freq = __expf(-0.28782313662f * (float)d);
  float ang  = (float)s * freq;
  float c  = __cosf(ang);
  float sn = __sinf(ang);

  float q1r = q1 * c - q2 * sn, q2r = q2 * c + q1 * sn;
  float k1r = k1 * c - k2 * sn, k2r = k2 * c + k1 * sn;

  size_t ro = (size_t)r * 64;
  qh[ro + d] = (_Float16)q1r;  qh[ro + d + 32] = (_Float16)q2r;
  kh[ro + d] = (_Float16)k1r;  kh[ro + d + 32] = (_Float16)k2r;
  vh[ro + d] = (_Float16)v1;   vh[ro + d + 32] = (_Float16)v2;
  kout[ro + d] = k1r;  kout[ro + d + 32] = k2r;
  vout[ro + d] = v1;   vout[ro + d + 32] = v2;
}

// --------------------------------------------------- flash attention (causal)
#define QB  128
#define KTL 32
#define DHD 64
#define VLD 40

__global__ __launch_bounds__(256) void k_attn(const _Float16* __restrict__ qh,
                                              const _Float16* __restrict__ kh,
                                              const _Float16* __restrict__ vh,
                                              _Float16* __restrict__ ctx,
                                              int S, int H) {
  __shared__ __align__(16) _Float16 Qs[QB * DHD];        // 16 KB
  __shared__ __align__(16) _Float16 Ks[KTL * DHD];       // 4 KB
  __shared__ __align__(16) _Float16 Vt[DHD * VLD];       // 5 KB (transposed V)
  __shared__ __align__(16) _Float16 Ps[8 * 16 * KTL];    // 8 KB (per-wave P)

  const int tid  = threadIdx.x;
  const int lane = tid & 31;
  const int wave = tid >> 5;
  const int l15  = lane & 15;
  const int lh   = lane >> 4;
  const int q0   = blockIdx.x * QB;
  const int bh   = blockIdx.y;            // b*H + h
  const int b    = bh / H, h = bh % H;

  const size_t rowbase = (size_t)bh * S;
  const _Float16* qsrc = qh + (rowbase + q0) * DHD;

  // Q tile (contiguous 128x64 halves = 16 KB) via async copy, 4 chunks/thread
#pragma unroll
  for (int c = 0; c < 4; ++c)
    async_copy16(&Qs[(tid + c * 256) * 8], qsrc + (tid + c * 256) * 8);
  async_wait0();
  __syncthreads();

  // Q A-fragments for this wave's 16 rows (two dh halves)
  v16h aq[2];
#pragma unroll
  for (int j = 0; j < 2; ++j) {
    const _Float16* p = &Qs[(wave * 16 + l15) * DHD + j * 32 + lh * 8];
    aq[j] = pack16(*(const v8h*)p, *(const v8h*)(p + 16));
  }

  v8f cacc[4] = {};
  float mrow[8], lrow[8];
#pragma unroll
  for (int v = 0; v < 8; ++v) { mrow[v] = -3.0e38f; lrow[v] = 0.0f; }

  const int nkt = q0 / KTL + QB / KTL;
  for (int kt = 0; kt < nkt; ++kt) {
    // K tile: contiguous 32x64 halves = 4 KB, async, 1 chunk/thread
    async_copy16(&Ks[tid * 8],
                 kh + (rowbase + (size_t)kt * KTL) * DHD + tid * 8);
    // V tile, stored transposed: Vt[d][k] (manual: converts layout in-flight)
    {
      const _Float16* vsrc = vh + (rowbase + (size_t)kt * KTL) * DHD;
      int k  = tid >> 3;          // 0..31
      int d0 = (tid & 7) * 8;     // 0..56
      v8h vv = *(const v8h*)(vsrc + k * DHD + d0);
#pragma unroll
      for (int j = 0; j < 8; ++j) Vt[(d0 + j) * VLD + k] = vv[j];
    }
    async_wait0();
    __syncthreads();

    // scores: q(16xdh) * K^T -> two 16x16 accumulators (32 keys)
    v8f sacc[2] = {};
#pragma unroll
    for (int g = 0; g < 2; ++g) {
#pragma unroll
      for (int j = 0; j < 2; ++j) {
        const _Float16* p = &Ks[(g * 16 + l15) * DHD + j * 32 + lh * 8];
        v16h bf = pack16(*(const v8h*)p, *(const v8h*)(p + 16));
        sacc[g] = wmma_f16(aq[j], bf, sacc[g]);
      }
    }

    const bool needMask = (kt * KTL + KTL - 1) > q0;
    float sv[2][8], rmax[8];
#pragma unroll
    for (int v = 0; v < 8; ++v) {
      int m_g = q0 + wave * 16 + lh * 8 + v;
      float s0 = sacc[0][v] * 0.125f;   // 1/sqrt(64)
      float s1 = sacc[1][v] * 0.125f;
      if (needMask) {
        if (kt * KTL + l15 > m_g)      s0 = -3.0e38f;
        if (kt * KTL + 16 + l15 > m_g) s1 = -3.0e38f;
      }
      sv[0][v] = s0; sv[1][v] = s1;
      float r = fmaxf(s0, s1);
      r = fmaxf(r, __shfl_xor(r, 1));
      r = fmaxf(r, __shfl_xor(r, 2));
      r = fmaxf(r, __shfl_xor(r, 4));
      r = fmaxf(r, __shfl_xor(r, 8));  // stays within each 16-lane half
      rmax[v] = r;
    }

#pragma unroll
    for (int v = 0; v < 8; ++v) {
      float mn    = fmaxf(mrow[v], rmax[v]);
      float alpha = __expf(mrow[v] - mn);
      mrow[v] = mn;
      float p0 = __expf(sv[0][v] - mn);
      float p1 = __expf(sv[1][v] - mn);
      float ps = p0 + p1;
      ps += __shfl_xor(ps, 1);
      ps += __shfl_xor(ps, 2);
      ps += __shfl_xor(ps, 4);
      ps += __shfl_xor(ps, 8);
      lrow[v] = lrow[v] * alpha + ps;
#pragma unroll
      for (int t = 0; t < 4; ++t) cacc[t][v] *= alpha;
      // stage P into per-wave LDS buffer: Ps[wave][m][n], m = 8*lh + v
      Ps[(wave * 16 + lh * 8 + v) * KTL + l15]      = (_Float16)p0;
      Ps[(wave * 16 + lh * 8 + v) * KTL + 16 + l15] = (_Float16)p1;
    }

    // P A-fragment (16x32) — wave-internal LDS ops are in-order
    const _Float16* pp = &Ps[(wave * 16 + l15) * KTL + lh * 8];
    v16h pf = pack16(*(const v8h*)pp, *(const v8h*)(pp + 16));

#pragma unroll
    for (int t = 0; t < 4; ++t) {
      const _Float16* pv = &Vt[(t * 16 + l15) * VLD + lh * 8];
      v16h bv = pack16(*(const v8h*)pv, *(const v8h*)(pv + 16));
      cacc[t] = wmma_f16(pf, bv, cacc[t]);
    }

    __syncthreads();
  }

  float inv[8];
#pragma unroll
  for (int v = 0; v < 8; ++v) inv[v] = 1.0f / lrow[v];

  // ctx stored directly in [B, S, H*dh] layout for the output GEMM
#pragma unroll
  for (int t = 0; t < 4; ++t)
#pragma unroll
    for (int v = 0; v < 8; ++v) {
      int m = q0 + wave * 16 + lh * 8 + v;
      int d = t * 16 + l15;
      ctx[(size_t)(b * S + m) * 1024 + h * 64 + d] =
          (_Float16)(cacc[t][v] * inv[v]);
    }
}

// ---------------------------------------------------------------------- host
extern "C" void kernel_launch(void* const* d_in, const int* in_sizes, int n_in,
                              void* d_out, int out_size, void* d_ws, size_t ws_size,
                              hipStream_t stream) {
  (void)in_sizes; (void)n_in; (void)out_size; (void)ws_size;
  constexpr int B = 2, S = 2048, D = 1024, H = 16;
  constexpr size_t MB = 1024 * 1024;

  const float* x   = (const float*)d_in[0];
  const float* qkv = (const float*)d_in[1];
  const float* wo  = (const float*)d_in[2];
  float* out = (float*)d_out;

  char* ws = (char*)d_ws;
  _Float16* xh   = (_Float16*)(ws + 0);        //  8 MB  x f16
  _Float16* qkvh = (_Float16*)(ws + 8  * MB);  //  6 MB  qkv f16
  _Float16* woh  = (_Float16*)(ws + 14 * MB);  //  2 MB  wo f16
  float*    QKV  = (float*)   (ws + 16 * MB);  // 48 MB  QKV f32
  _Float16* qhp  = (_Float16*)(ws + 64 * MB);  //  8 MB  q rope f16 [B,H,S,dh]
  _Float16* khp  = (_Float16*)(ws + 72 * MB);  //  8 MB  k rope f16
  _Float16* vhp  = (_Float16*)(ws + 80 * MB);  //  8 MB  v f16
  _Float16* ctxh = (_Float16*)(ws + 88 * MB);  //  8 MB  ctx f16 [B,S,D]

  float* kout = out + (size_t)B * S * D;            // k output (f32)
  float* vout = kout + (size_t)B * H * S * (D / H); // v output (f32)

  k_cvt<<<dim3((B * S * D) / 1024), 256, 0, stream>>>(x, xh, B * S * D);
  k_cvt<<<dim3((3 * D * D) / 1024), 256, 0, stream>>>(qkv, qkvh, 3 * D * D);
  k_cvt<<<dim3((D * D) / 1024), 256, 0, stream>>>(wo, woh, D * D);

  k_gemm<3 * D, D><<<dim3((B * S) / GBM, (3 * D) / GBN), 256, 0, stream>>>(
      xh, qkvh, QKV);
  k_rope<<<dim3((B * H * S) / 8), 256, 0, stream>>>(
      QKV, qhp, khp, vhp, kout, vout, S, H);
  k_attn<<<dim3(S / QB, B * H), 256, 0, stream>>>(qhp, khp, vhp, ctxh, S, H);
  k_gemm<D, D><<<dim3((B * S) / GBM, D / GBN), 256, 0, stream>>>(
      ctxh, woh, out);
}
